// VisualDict_26079041422083
// MI455X (gfx1250) — matrix operand order repaired and checked by reading, not compiled, not run on hardware
//
#include <hip/hip_runtime.h>

#define N_TOK  16384
#define K_CODE 8192
#define D_DIM  256
#define CHUNK  32              // codebook rows per LDS buffer
#define WAVES  4               // waves per block -> 64 token rows per block
#define LDS_STRIDE (D_DIM + 8) // pad 16B per row to spread LDS banks

typedef __attribute__((ext_vector_type(16))) __bf16       v16bf;
typedef __attribute__((ext_vector_type(8)))  float        v8f;
typedef __attribute__((ext_vector_type(4)))  unsigned int v4u;
typedef __attribute__((ext_vector_type(4)))  int          v4i;

union Frag16 { v16bf v; v4u u[2]; };

// ---- CDNA5 async global->LDS copy (16B per lane), ASYNCcnt tracked ----
__device__ __forceinline__ void async_copy16(void* lds_dst, const void* gsrc) {
#if __has_builtin(__builtin_amdgcn_global_load_async_to_lds_b128)
  __builtin_amdgcn_global_load_async_to_lds_b128(
      (__attribute__((address_space(1))) v4i*)gsrc,
      (__attribute__((address_space(3))) v4i*)lds_dst, 0, 0);
#else
  unsigned int loff =
      (unsigned int)(unsigned long long)(__attribute__((address_space(3))) char*)lds_dst;
  unsigned long long ga = (unsigned long long)gsrc;
  asm volatile("global_load_async_to_lds_b128 %0, %1, off"
               :: "v"(loff), "v"(ga) : "memory");
#endif
}

__device__ __forceinline__ void async_wait0() {
#if __has_builtin(__builtin_amdgcn_s_wait_asynccnt)
  __builtin_amdgcn_s_wait_asynccnt(0);
#else
  asm volatile("s_wait_asynccnt 0x0" ::: "memory");
#endif
}

__device__ inline unsigned short f32_to_bf16_rn(float f) {
  unsigned int u = __float_as_uint(f);
  u += 0x7FFFu + ((u >> 16) & 1u);
  return (unsigned short)(u >> 16);
}

// fp32 -> (bf16 hi, bf16 lo) split:  x ~= hi + lo, lo = rn(x - float(hi))
__global__ __launch_bounds__(256) void split_kernel(const float* __restrict__ in,
                                                    unsigned short* __restrict__ hi,
                                                    unsigned short* __restrict__ lo,
                                                    int n) {
  int i = blockIdx.x * blockDim.x + threadIdx.x;
  if (i >= n) return;
  float x = in[i];
  unsigned short hb = f32_to_bf16_rn(x);
  float hf = __uint_as_float(((unsigned int)hb) << 16);
  hi[i] = hb;
  lo[i] = f32_to_bf16_rn(x - hf);
}

// ||e_k||^2 in fp32; 8 lanes per row, 32 rows per 256-thread block
__global__ __launch_bounds__(256) void rowsq_kernel(const float* __restrict__ e,
                                                    float* __restrict__ esq) {
  const int t    = threadIdx.x;
  const int row  = blockIdx.x * 32 + (t >> 3);
  const int slot = t & 7;
  const float4* p = (const float4*)(e + (size_t)row * D_DIM);
  float s = 0.f;
  for (int i = slot; i < D_DIM / 4; i += 8) {
    float4 v = p[i];
    s += v.x * v.x + v.y * v.y + v.z * v.z + v.w * v.w;
  }
  s += __shfl_xor(s, 1, 32);
  s += __shfl_xor(s, 2, 32);
  s += __shfl_xor(s, 4, 32);
  if (slot == 0) esq[row] = s;
}

// async-DMA one CHUNK of codebook rows (hi+lo) into an LDS buffer
__device__ __forceinline__ void stage_chunk(
    const unsigned short* __restrict__ ehi, const unsigned short* __restrict__ elo,
    unsigned short (*dsthi)[LDS_STRIDE], unsigned short (*dstlo)[LDS_STRIDE],
    int c0, int tid) {
  for (int t = tid; t < CHUNK * (D_DIM / 8); t += (WAVES * 32)) {
    const int r = t / (D_DIM / 8);
    const int c = (t % (D_DIM / 8)) * 8;
    async_copy16(&dsthi[r][c], ehi + (size_t)(c0 + r) * D_DIM + c);
    async_copy16(&dstlo[r][c], elo + (size_t)(c0 + r) * D_DIM + c);
  }
}

// Core: per wave, 16 token rows vs all 8192 codes.
// score = e_sq[k] - 2 * x.e_k   via bf16x3 WMMA (hi*hi + hi*lo + lo*hi)
__global__ __launch_bounds__(WAVES * 32) void vq_argmin_kernel(
    const unsigned short* __restrict__ xhi, const unsigned short* __restrict__ xlo,
    const unsigned short* __restrict__ ehi, const unsigned short* __restrict__ elo,
    const float* __restrict__ esq, int* __restrict__ idx_out) {
  __shared__ unsigned short s_hi[2][CHUNK][LDS_STRIDE];
  __shared__ unsigned short s_lo[2][CHUNK][LDS_STRIDE];

  const int lane = threadIdx.x & 31;
  const int wave = threadIdx.x >> 5;
  const int row_base = blockIdx.x * (WAVES * 16) + wave * 16;

  const int arow = row_base + (lane & 15);
  const int ka = (lane >> 4) * 8;   // A: halves {0..7} at ka, {8..15} at ka+16 per 32-k step
  const int kb = (lane >> 4) * 8;   // B: halves {0..7} at kb, {8..15} at kb+16 per 32-k step

  // kick off DMA of chunk 0 before touching A (overlap with A-fragment loads)
  stage_chunk(ehi, elo, s_hi[0], s_lo[0], 0, threadIdx.x);

  // A tile (16 rows x 256 d, hi+lo) resident in VGPRs for the whole K sweep
  v16bf ahi[8], alo[8];
  {
    const unsigned short* xh = xhi + (size_t)arow * D_DIM;
    const unsigned short* xl = xlo + (size_t)arow * D_DIM;
#pragma unroll
    for (int ks = 0; ks < 8; ++ks) {
      Frag16 fh, fl;
      const int k0 = ks * 32 + ka;
      fh.u[0] = *(const v4u*)(xh + k0);
      fh.u[1] = *(const v4u*)(xh + k0 + 16);
      fl.u[0] = *(const v4u*)(xl + k0);
      fl.u[1] = *(const v4u*)(xl + k0 + 16);
      ahi[ks] = fh.v;
      alo[ks] = fl.v;
    }
  }

  float minv[8];
  int   mini[8];
#pragma unroll
  for (int v = 0; v < 8; ++v) { minv[v] = 3.4e38f; mini[v] = 0; }

  async_wait0();
  __syncthreads();

  for (int c0 = 0, it = 0; c0 < K_CODE; c0 += CHUNK, ++it) {
    const int buf = it & 1;
    // prefetch next chunk into the other buffer while we compute on this one
    if (c0 + CHUNK < K_CODE)
      stage_chunk(ehi, elo, s_hi[buf ^ 1], s_lo[buf ^ 1], c0 + CHUNK, threadIdx.x);

    for (int sub = 0; sub < CHUNK / 16; ++sub) {
      const int nloc = sub * 16 + (lane & 15);
      const unsigned short* bh = &s_hi[buf][nloc][0];
      const unsigned short* bl = &s_lo[buf][nloc][0];
      v8f acc0 = {};
      v8f acc1 = {};
#pragma unroll
      for (int ks = 0; ks < 8; ++ks) {
        Frag16 fbh, fbl;
        const int k0 = ks * 32 + kb;
        fbh.u[0] = *(const v4u*)(bh + k0);
        fbh.u[1] = *(const v4u*)(bh + k0 + 16);
        fbl.u[0] = *(const v4u*)(bl + k0);
        fbl.u[1] = *(const v4u*)(bl + k0 + 16);
        v8f* ap = (ks & 1) ? &acc1 : &acc0;  // two chains to keep the matrix pipe busy
        *ap = __builtin_amdgcn_wmma_f32_16x16x32_bf16(false, alo[ks], false, fbh.v,
                                                      (short)0, *ap, false, false);
        *ap = __builtin_amdgcn_wmma_f32_16x16x32_bf16(false, ahi[ks], false, fbl.v,
                                                      (short)0, *ap, false, false);
        *ap = __builtin_amdgcn_wmma_f32_16x16x32_bf16(false, ahi[ks], false, fbh.v,
                                                      (short)0, *ap, false, false);
      }
      const float es = esq[c0 + nloc];  // L2-resident, coalesced 64B per half-wave
      const int n = c0 + nloc;
#pragma unroll
      for (int v = 0; v < 8; ++v) {
        const float sc = es - 2.0f * (acc0[v] + acc1[v]);
        if (sc < minv[v]) { minv[v] = sc; mini[v] = n; }
      }
    }

    async_wait0();    // next-chunk DMA landed in LDS
    __syncthreads();  // all waves done reading this buffer before it's overwritten
  }

  // per-row argmin across the 16 column-lanes of each half-wave (+index tiebreak)
#pragma unroll
  for (int v = 0; v < 8; ++v) {
    float m = minv[v];
    int   mi = mini[v];
#pragma unroll
    for (int off = 8; off >= 1; off >>= 1) {
      const float om = __shfl_xor(m, off, 32);
      const int   oi = __shfl_xor(mi, off, 32);
      if (om < m || (om == m && oi < mi)) { m = om; mi = oi; }
    }
    if (lane == 0)  idx_out[row_base + v]     = mi;  // rows 0..7  live in lanes 0-15
    if (lane == 16) idx_out[row_base + 8 + v] = mi;  // rows 8..15 live in lanes 16-31
  }
}

// quantize[n, :] = embed[idx[n], :] in exact fp32; also emit indices
__global__ __launch_bounds__(64) void gather_kernel(const float* __restrict__ embed,
                                                    const int* __restrict__ idx,
                                                    float* __restrict__ qout,
                                                    int* __restrict__ iout) {
  const int n = blockIdx.x;
  const int id = idx[n];
  const float4* src = (const float4*)(embed + (size_t)id * D_DIM);
  float4* dst = (float4*)(qout + (size_t)n * D_DIM);
  for (int i = threadIdx.x; i < D_DIM / 4; i += 64) dst[i] = src[i];
  if (threadIdx.x == 0) iout[n] = id;
}

extern "C" void kernel_launch(void* const* d_in, const int* in_sizes, int n_in,
                              void* d_out, int out_size, void* d_ws, size_t ws_size,
                              hipStream_t stream) {
  (void)in_sizes; (void)n_in; (void)out_size; (void)ws_size;
  const float* x = (const float*)d_in[0];  // [N, D] fp32
  const float* e = (const float*)d_in[1];  // [K, D] fp32

  // workspace: x hi/lo (8+8 MB), e hi/lo (4+4 MB), e_sq (32 KB), idx (64 KB)
  unsigned short* xhi = (unsigned short*)d_ws;
  unsigned short* xlo = xhi + (size_t)N_TOK * D_DIM;
  unsigned short* ehi = xlo + (size_t)N_TOK * D_DIM;
  unsigned short* elo = ehi + (size_t)K_CODE * D_DIM;
  float* esq = (float*)(elo + (size_t)K_CODE * D_DIM);
  int*   idx = (int*)(esq + K_CODE);

  float* qout = (float*)d_out;                         // [N, D] fp32
  int*   iout = (int*)(qout + (size_t)N_TOK * D_DIM);  // [N] indices

  split_kernel<<<(N_TOK * D_DIM) / 256, 256, 0, stream>>>(x, xhi, xlo, N_TOK * D_DIM);
  split_kernel<<<(K_CODE * D_DIM) / 256, 256, 0, stream>>>(e, ehi, elo, K_CODE * D_DIM);
  rowsq_kernel<<<K_CODE / 32, 256, 0, stream>>>(e, esq);
  vq_argmin_kernel<<<N_TOK / (WAVES * 16), WAVES * 32, 0, stream>>>(xhi, xlo, ehi, elo,
                                                                    esq, idx);
  gather_kernel<<<N_TOK, 64, 0, stream>>>(e, idx, qout, iout);
}
